// MGSAParallelEncoder_644245095063
// MI455X (gfx1250) — matrix operand order repaired and checked by reading, not compiled, not run on hardware
//
#include <hip/hip_runtime.h>
#include <hip/hip_bf16.h>
#include <math.h>

#define TT 2048
#define DD 1024
#define HH 16
#define DH 64

// ---------- WMMA types / helpers (gfx1250, wave32) ----------
typedef __bf16 v16bf __attribute__((ext_vector_type(16)));
typedef float  v8f   __attribute__((ext_vector_type(8)));
typedef unsigned int v4u __attribute__((ext_vector_type(4)));
typedef int v8i __attribute__((ext_vector_type(8)));
typedef int v4i __attribute__((ext_vector_type(4)));

union Frag16 { v16bf v; v4u q[2]; unsigned short u[16]; };

static __device__ __forceinline__ unsigned short f2bf(float f) {
    unsigned int u = __float_as_uint(f);
    u += 0x7FFFu + ((u >> 16) & 1u);          // round-to-nearest-even
    return (unsigned short)(u >> 16);
}
static __device__ __forceinline__ float bf2f(unsigned short s) {
    return __uint_as_float(((unsigned int)s) << 16);
}

// A-fragment (16x32 bf16, M x K). Row-major source, lda in elements. 2 x b128 per lane.
static __device__ __forceinline__ v16bf load_a16(const unsigned short* A, int lda, int row0, int k0) {
    int lane = threadIdx.x & 31;
    int m = lane & 15, hi = lane >> 4;
    const v4u* p = (const v4u*)(A + (size_t)(row0 + m) * lda + k0 + hi * 8);
    Frag16 f; f.q[0] = p[0]; f.q[1] = p[2];
    return f.v;
}
// B-fragment (32x16 bf16, K x N) from a ROW-MAJOR N x K ("transposed") source. 2 x b128 per lane.
static __device__ __forceinline__ v16bf load_bT16(const unsigned short* Bt, int ldb, int k0, int n0) {
    int lane = threadIdx.x & 31;
    int n = lane & 15, hi = lane >> 4;
    const v4u* p = (const v4u*)(Bt + (size_t)(n0 + n) * ldb + k0 + hi * 8);
    Frag16 f; f.q[0] = p[0]; f.q[1] = p[2];
    return f.v;
}
static __device__ __forceinline__ v8f wmma_bf16(v16bf a, v16bf b, v8f c) {
    return __builtin_amdgcn_wmma_f32_16x16x32_bf16(false, a, false, b, (short)0, c, false, false);
}

// ---------- Tensor Data Mover: 2D tile (bf16) global -> LDS ----------
// Builds the D# descriptor groups per CDNA5 ISA 8.3-8.6.
// tileX = contiguous elements per row, tileY = rows, strideX = row stride (elements).
// Tensor dims == tile dims (tile always in-bounds).
static __device__ __forceinline__ void tdm_load_2d(unsigned lds_off, const void* gptr,
                                                   unsigned tileX, unsigned tileY,
                                                   unsigned long long strideX)
{
    unsigned long long ga = (unsigned long long)(size_t)gptr;
    v4u g0 = { 1u,                                    // count=1, user mode
               lds_off,                               // lds_addr (bytes)
               (unsigned)(ga & 0xFFFFFFFFu),          // global_addr[31:0]
               ((unsigned)((ga >> 32) & 0x01FFFFFFu)) | 0x80000000u }; // addr[56:32] | type=2
    v8i g1 = { (int)0x10000,                                          // data_size=1 (2B), wg_mask=0
               (int)((tileX & 0xFFFFu) << 16),                        // tensor_dim0[15:0]
               (int)(((tileX >> 16) & 0xFFFFu) | ((tileY & 0xFFFFu) << 16)),
               (int)(((tileY >> 16) & 0xFFFFu) | ((tileX & 0xFFFFu) << 16)), // tile_dim0
               (int)(tileY & 0xFFFFu),                                // tile_dim1 (tile_dim2=0)
               (int)(strideX & 0xFFFFFFFFull),                        // tensor_dim0_stride lo
               (int)((strideX >> 32) & 0xFFFFull),                    // stride hi | dim1_stride=0
               0 };
    v4i g2 = { 0, 0, 0, 0 };
    v4i g3 = { 0, 0, 0, 0 };
#if defined(__clang_major__) && (__clang_major__ >= 23)
    v8i g4 = { 0, 0, 0, 0, 0, 0, 0, 0 };
    __builtin_amdgcn_tensor_load_to_lds(g0, g1, g2, g3, g4, 0);
#else
    __builtin_amdgcn_tensor_load_to_lds(g0, g1, g2, g3, 0);
#endif
}

// ---------- conversion kernels ----------
__global__ void cvt_kernel(const float* __restrict__ s, unsigned short* __restrict__ d, int n) {
    int i = blockIdx.x * 256 + threadIdx.x;
    if (i < n) d[i] = f2bf(s[i]);
}
__global__ void cvtT_kernel(const float* __restrict__ s, unsigned short* __restrict__ d, int K, int N) {
    int i = blockIdx.x * 256 + threadIdx.x;
    if (i < K * N) {
        int k = i / N, n = i % N;
        d[(size_t)n * K + k] = f2bf(s[i]);
    }
}
__global__ void diag_kernel(const float* __restrict__ ld, float* __restrict__ dg, int n) {
    int i = blockIdx.x * 256 + threadIdx.x;
    if (i < n) {
        float x = ld[i];
        float sp = (x > 0.f) ? (x + log1pf(__expf(-x))) : log1pf(__expf(x));
        dg[i] = sp + 1e-6f;
    }
}

// ---------- WMMA GEMM: 32x64 tile per wave (2x4 accumulators) ----------
// C[M x N] = A[M x K](bf16) @ Wt(N x K bf16, transposed) + bias
// epi: 0 = bf16 row-major, 1 = bf16 TRANSPOSED (col*M+row), 2 = gelu->bf16, 3 = f32 row-major
__global__ __launch_bounds__(128) void gemm_bf16_kernel(
    const unsigned short* __restrict__ A, const unsigned short* __restrict__ Wt,
    const float* __restrict__ bias, unsigned short* __restrict__ out16,
    float* __restrict__ out32, int M, int N, int K, int epi)
{
    int wv = threadIdx.x >> 5;
    int n0 = blockIdx.x * 256 + wv * 64;
    int row0 = blockIdx.y * 32;

    v8f acc[2][4];
#pragma unroll
    for (int ri = 0; ri < 2; ri++)
#pragma unroll
        for (int nt = 0; nt < 4; nt++)
            acc[ri][nt] = (v8f){0.f,0.f,0.f,0.f,0.f,0.f,0.f,0.f};

    for (int k0 = 0; k0 < K; k0 += 32) {
        v16bf a0 = load_a16(A, K, row0,      k0);
        v16bf a1 = load_a16(A, K, row0 + 16, k0);
#pragma unroll
        for (int nt = 0; nt < 4; nt++) {
            v16bf b = load_bT16(Wt, K, k0, n0 + nt * 16);
            acc[0][nt] = wmma_bf16(a0, b, acc[0][nt]);
            acc[1][nt] = wmma_bf16(a1, b, acc[1][nt]);
        }
    }

    int lane = threadIdx.x & 31;
    int nn = lane & 15, hi = lane >> 4;
#pragma unroll
    for (int ri = 0; ri < 2; ri++) {
#pragma unroll
        for (int nt = 0; nt < 4; nt++) {
            int col = n0 + nt * 16 + nn;
            float bv = bias[col];
#pragma unroll
            for (int r = 0; r < 8; r++) {
                int row = row0 + ri * 16 + r + hi * 8;
                float x = acc[ri][nt][r] + bv;
                if (epi == 0) {
                    out16[(size_t)row * N + col] = f2bf(x);
                } else if (epi == 1) {
                    out16[(size_t)col * M + row] = f2bf(x);
                } else if (epi == 2) {
                    float g = 0.5f * x * (1.f + erff(x * 0.70710678f));
                    out16[(size_t)row * N + col] = f2bf(g);
                } else {
                    out32[(size_t)row * N + col] = x;
                }
            }
        }
    }
}

// ---------- qM = qg*diag, q2 = sum qg^2*diag, k2 = sum kg^2*diag ----------
__global__ __launch_bounds__(32) void qm_kernel(
    const unsigned short* __restrict__ qg, const unsigned short* __restrict__ kg,
    const float* __restrict__ diag, unsigned short* __restrict__ qM,
    float* __restrict__ q2, float* __restrict__ k2)
{
    int t = blockIdx.x, h = blockIdx.y, lane = threadIdx.x;
    float sq = 0.f, sk = 0.f;
#pragma unroll
    for (int rep = 0; rep < 2; rep++) {
        int dloc = lane + rep * 32;
        size_t idx = (size_t)t * DD + h * DH + dloc;
        float dg = diag[h * DH + dloc];
        float q = bf2f(qg[idx]);
        float qm = q * dg;
        qM[idx] = f2bf(qm);
        sq += q * qm;
        float kk = bf2f(kg[idx]);
        sk += kk * kk * dg;
    }
    for (int off = 16; off; off >>= 1) { sq += __shfl_xor(sq, off); sk += __shfl_xor(sk, off); }
    if (lane == 0) { q2[h * TT + t] = sq; k2[h * TT + t] = sk; }
}

// ---------- flash-attention, TDM double-buffered K/V staging ----------
// 1 wave owns (head h, 16-row tile). K tile: 32 keys x 64 dh (row-major from K).
// V tile: 64 dh x 32 keys (row-major from V^T). Both DMA'd by the Tensor Data Mover.
__global__ __launch_bounds__(32) void attn_kernel(
    const unsigned short* __restrict__ Qb, const unsigned short* __restrict__ Kb,
    const unsigned short* __restrict__ Vt,   // V transposed: [D x T] bf16
    const unsigned char* __restrict__ mask,
    const float* __restrict__ q2, const float* __restrict__ k2,
    unsigned short* __restrict__ outAtt, int isGsa)
{
    __shared__ __attribute__((aligned(16))) unsigned short Kt[2][32 * 64];
    __shared__ __attribute__((aligned(16))) unsigned short Vl[2][64 * 32];
    __shared__ __attribute__((aligned(16))) float Sld[16 * 32];
    __shared__ __attribute__((aligned(16))) unsigned short Pld[16 * 32];
    __shared__ float rowM[16], rowL[16], rowA[16];

    int h = blockIdx.y;
    int t0 = blockIdx.x * 16;
    int lane = threadIdx.x & 31;
    int nn = lane & 15, hi = lane >> 4;

    const unsigned short* Qh = Qb + h * DH;                 // row stride DD
    const unsigned short* Kh = Kb + h * DH;                 // row stride DD
    const unsigned short* Vh = Vt + (size_t)(h * DH) * TT;  // [Dh x T]

    v8f acc[4];
#pragma unroll
    for (int nt = 0; nt < 4; nt++) acc[nt] = (v8f){0.f,0.f,0.f,0.f,0.f,0.f,0.f,0.f};

    if (lane < 16) { rowM[lane] = -1e30f; rowL[lane] = 0.f; }
    __syncthreads();

    const float cM = 0.125f;                  // 1/sqrt(64)
    const float cG = -0.5f * 0.125f * 0.25f;  // -0.5*inv_sqrt_dh*GSA_SCALE

    // Loop-invariant Q fragments and per-row GSA q2 terms
    v16bf qa0 = load_a16(Qh, DD, t0, 0);
    v16bf qa1 = load_a16(Qh, DD, t0, 32);
    float q2r[8];
#pragma unroll
    for (int r = 0; r < 8; r++) q2r[r] = isGsa ? q2[h * TT + t0 + r + hi * 8] : 0.f;

    unsigned kOff0 = (unsigned)(size_t)(void*)&Kt[0][0];
    unsigned kOff1 = (unsigned)(size_t)(void*)&Kt[1][0];
    unsigned vOff0 = (unsigned)(size_t)(void*)&Vl[0][0];
    unsigned vOff1 = (unsigned)(size_t)(void*)&Vl[1][0];

    // prologue: DMA first K/V tiles into buffer 0
    tdm_load_2d(kOff0, Kh, 64, 32, (unsigned long long)DD);
    tdm_load_2d(vOff0, Vh, 32, 64, (unsigned long long)TT);

    for (int s0 = 0; s0 < TT; s0 += 32) {
        int buf = (s0 >> 5) & 1;
        if (s0 + 32 < TT) {
            // DMA next tiles into the other buffer, then wait for current buffer only
            tdm_load_2d(buf ? kOff0 : kOff1, Kh + (size_t)(s0 + 32) * DD, 64, 32, (unsigned long long)DD);
            tdm_load_2d(buf ? vOff0 : vOff1, Vh + (s0 + 32), 32, 64, (unsigned long long)TT);
            __builtin_amdgcn_s_wait_tensorcnt((short)2);
        } else {
            __builtin_amdgcn_s_wait_tensorcnt((short)0);
        }
        const unsigned short* Kl = &Kt[buf][0];   // [32 keys][64 dh]
        const unsigned short* Vb = &Vl[buf][0];   // [64 dh][32 keys]

        // ---- scores: two 16x16 fragments from LDS K tile ----
#pragma unroll
        for (int half = 0; half < 2; half++) {
            v8f cs = {0.f,0.f,0.f,0.f,0.f,0.f,0.f,0.f};
            cs = wmma_bf16(qa0, load_bT16(Kl, 64, 0,  half * 16), cs);
            cs = wmma_bf16(qa1, load_bT16(Kl, 64, 32, half * 16), cs);
            int colG = s0 + half * 16 + nn;
            float k2c = isGsa ? k2[h * TT + colG] : 0.f;
#pragma unroll
            for (int r = 0; r < 8; r++) {
                int rr = r + hi * 8;
                int tg = t0 + rr;
                float sv;
                if (isGsa) sv = cG * (q2r[r] + k2c - 2.f * cs[r]);
                else       sv = cs[r] * cM;
                if (!mask[(size_t)tg * TT + colG]) sv = -1e30f;
                Sld[rr * 32 + half * 16 + nn] = sv;
            }
        }
        __syncthreads();

        // ---- online softmax row stats (lanes 0..15, one row each) ----
        if (lane < 16) {
            int rr = lane;
            float mOld = rowM[rr];
            float bm = mOld;
            for (int j = 0; j < 32; j++) bm = fmaxf(bm, Sld[rr * 32 + j]);
            float al = __expf(mOld - bm);
            float ls = 0.f;
            for (int j = 0; j < 32; j++) {
                float p = __expf(Sld[rr * 32 + j] - bm);
                Pld[rr * 32 + j] = f2bf(p);
                ls += p;
            }
            rowM[rr] = bm;
            rowA[rr] = al;
            rowL[rr] = rowL[rr] * al + ls;
        }
        __syncthreads();

        // ---- rescale accumulators + P@V (A from LDS P, B from LDS V tile) ----
        v16bf pa = load_a16((const unsigned short*)Pld, 32, 0, 0);
#pragma unroll
        for (int nt = 0; nt < 4; nt++) {
#pragma unroll
            for (int r = 0; r < 8; r++) acc[nt][r] *= rowA[r + hi * 8];
            v16bf vb = load_bT16(Vb, 32, 0, nt * 16);
            acc[nt] = wmma_bf16(pa, vb, acc[nt]);
        }
        __syncthreads();
    }

    // ---- normalize + store bf16 (row-major [T x D], head slice) ----
#pragma unroll
    for (int nt = 0; nt < 4; nt++) {
#pragma unroll
        for (int r = 0; r < 8; r++) {
            int rr = r + hi * 8;
            float l = rowL[rr];
            float v = (l > 0.f) ? acc[nt][r] / l : 0.f;
            outAtt[(size_t)(t0 + rr) * DD + h * DH + nt * 16 + nn] = f2bf(v);
        }
    }
}

// ---------- fused epilogue: gate scalar + two LayerNorms + blend ----------
__global__ __launch_bounds__(256) void final_kernel(
    const float* __restrict__ projM, const float* __restrict__ projG,
    const unsigned short* __restrict__ h1, const float* __restrict__ w2,
    const float* __restrict__ b2,
    const float* __restrict__ lnMg, const float* __restrict__ lnMb,
    const float* __restrict__ lnGg, const float* __restrict__ lnGb,
    const float* __restrict__ gsaMix, float* __restrict__ out)
{
    __shared__ float rA[256], rB[256];
    int t = blockIdx.x, tid = threadIdx.x;

    float p = 0.f;
    for (int j = tid; j < DD; j += 256) p += bf2f(h1[(size_t)t * DD + j]) * w2[j];
    rA[tid] = p; __syncthreads();
    for (int s = 128; s; s >>= 1) { if (tid < s) rA[tid] += rA[tid + s]; __syncthreads(); }
    float alpha = 0.9f / (1.f + __expf(-(rA[0] + b2[0])));
    __syncthreads();

    float s1 = 0.f, s2 = 0.f;
    for (int j = tid; j < DD; j += 256) { float x = projM[(size_t)t * DD + j]; s1 += x; s2 += x * x; }
    rA[tid] = s1; rB[tid] = s2; __syncthreads();
    for (int s = 128; s; s >>= 1) { if (tid < s) { rA[tid] += rA[tid + s]; rB[tid] += rB[tid + s]; } __syncthreads(); }
    float mM = rA[0] / DD;
    float vM = rB[0] / DD - mM * mM;
    float rsM = rsqrtf(vM + 1e-5f);
    __syncthreads();

    s1 = 0.f; s2 = 0.f;
    for (int j = tid; j < DD; j += 256) { float x = projG[(size_t)t * DD + j]; s1 += x; s2 += x * x; }
    rA[tid] = s1; rB[tid] = s2; __syncthreads();
    for (int s = 128; s; s >>= 1) { if (tid < s) { rA[tid] += rA[tid + s]; rB[tid] += rB[tid + s]; } __syncthreads(); }
    float mG = rA[0] / DD;
    float vG = rB[0] / DD - mG * mG;
    float rsG = rsqrtf(vG + 1e-5f);

    float sigmix = 1.f / (1.f + __expf(-gsaMix[0]));
    for (int j = tid; j < DD; j += 256) {
        float a = (projM[(size_t)t * DD + j] - mM) * rsM * lnMg[j] + lnMb[j];
        float g = (projG[(size_t)t * DD + j] - mG) * rsG * lnGg[j] + lnGb[j];
        out[(size_t)t * DD + j] = alpha * a + (1.f - alpha) * sigmix * g;
    }
}

// ---------- host orchestration ----------
extern "C" void kernel_launch(void* const* d_in, const int* in_sizes, int n_in,
                              void* d_out, int out_size, void* d_ws, size_t ws_size,
                              hipStream_t stream)
{
    (void)in_sizes; (void)n_in; (void)out_size; (void)ws_size;
    const float* x = (const float*)d_in[0];
    const unsigned char* mask = (const unsigned char*)d_in[1];

    char* ws = (char*)d_ws;
    size_t off = 0;
    auto alloc = [&](size_t bytes) -> void* {
        void* p = ws + off;
        off += (bytes + 255) & ~(size_t)255;
        return p;
    };

    const size_t TD2 = (size_t)TT * DD * 2;
    const size_t DDb = (size_t)DD * DD * 2;

    unsigned short* xb = (unsigned short*)alloc(TD2);
    const int widx[9] = {2, 4, 6, 8, 10, 12, 14, 16, 24};
    unsigned short* Wt[9];
    for (int i = 0; i < 9; i++) Wt[i] = (unsigned short*)alloc(DDb);

    unsigned short* qb  = (unsigned short*)alloc(TD2);
    unsigned short* kb  = (unsigned short*)alloc(TD2);
    unsigned short* vT  = (unsigned short*)alloc(TD2);  // [D x T]
    unsigned short* qgb = (unsigned short*)alloc(TD2);
    unsigned short* kgb = (unsigned short*)alloc(TD2);
    unsigned short* vgT = (unsigned short*)alloc(TD2);  // [D x T]
    unsigned short* qMb = (unsigned short*)alloc(TD2);
    unsigned short* h1b = (unsigned short*)alloc(TD2);
    unsigned short* attM = (unsigned short*)alloc(TD2);
    unsigned short* attG = (unsigned short*)alloc(TD2);
    float* diagf = (float*)alloc((size_t)HH * DH * 4);
    float* q2f   = (float*)alloc((size_t)HH * TT * 4);
    float* k2f   = (float*)alloc((size_t)HH * TT * 4);
    float* projM = (float*)alloc((size_t)TT * DD * 4);
    float* projG = (float*)alloc((size_t)TT * DD * 4);

    int nTD = TT * DD;
    cvt_kernel<<<(nTD + 255) / 256, 256, 0, stream>>>(x, xb, nTD);
    for (int i = 0; i < 9; i++)
        cvtT_kernel<<<(DD * DD + 255) / 256, 256, 0, stream>>>((const float*)d_in[widx[i]], Wt[i], DD, DD);
    diag_kernel<<<(HH * DH + 255) / 256, 256, 0, stream>>>((const float*)d_in[18], diagf, HH * DH);

    dim3 gg(DD / 256, TT / 32);
    gemm_bf16_kernel<<<gg, 128, 0, stream>>>(xb, Wt[0], (const float*)d_in[3],  qb,  nullptr, TT, DD, DD, 0);
    gemm_bf16_kernel<<<gg, 128, 0, stream>>>(xb, Wt[1], (const float*)d_in[5],  kb,  nullptr, TT, DD, DD, 0);
    gemm_bf16_kernel<<<gg, 128, 0, stream>>>(xb, Wt[2], (const float*)d_in[7],  vT,  nullptr, TT, DD, DD, 1);
    gemm_bf16_kernel<<<gg, 128, 0, stream>>>(xb, Wt[4], (const float*)d_in[11], qgb, nullptr, TT, DD, DD, 0);
    gemm_bf16_kernel<<<gg, 128, 0, stream>>>(xb, Wt[5], (const float*)d_in[13], kgb, nullptr, TT, DD, DD, 0);
    gemm_bf16_kernel<<<gg, 128, 0, stream>>>(xb, Wt[6], (const float*)d_in[15], vgT, nullptr, TT, DD, DD, 1);
    gemm_bf16_kernel<<<gg, 128, 0, stream>>>(xb, Wt[8], (const float*)d_in[25], h1b, nullptr, TT, DD, DD, 2);

    qm_kernel<<<dim3(TT, HH), 32, 0, stream>>>(qgb, kgb, diagf, qMb, q2f, k2f);

    attn_kernel<<<dim3(TT / 16, HH), 32, 0, stream>>>(qb,  kb,  vT,  mask, q2f, k2f, attM, 0);
    attn_kernel<<<dim3(TT / 16, HH), 32, 0, stream>>>(qMb, kgb, vgT, mask, q2f, k2f, attG, 1);

    gemm_bf16_kernel<<<gg, 128, 0, stream>>>(attM, Wt[3], (const float*)d_in[9],  nullptr, projM, TT, DD, DD, 3);
    gemm_bf16_kernel<<<gg, 128, 0, stream>>>(attG, Wt[7], (const float*)d_in[17], nullptr, projG, TT, DD, DD, 3);

    final_kernel<<<TT, 256, 0, stream>>>(projM, projG, h1b,
                                         (const float*)d_in[26], (const float*)d_in[27],
                                         (const float*)d_in[19], (const float*)d_in[20],
                                         (const float*)d_in[21], (const float*)d_in[22],
                                         (const float*)d_in[23], (float*)d_out);
}
